// residualBlock_58978490909280
// MI455X (gfx1250) — matrix operand order repaired and checked by reading, not compiled, not run on hardware
//
#include <hip/hip_runtime.h>
#include <hip/hip_bf16.h>

// ---------------------------------------------------------------------------
// Deformable-conv residual block for gfx1250 (MI455X).
// bf16 WMMA GEMM cores + async global->LDS weight staging (ASYNCcnt).
// ---------------------------------------------------------------------------

typedef __attribute__((ext_vector_type(16))) __bf16 v16bf;
typedef __attribute__((ext_vector_type(8)))  __bf16 v8bf;
typedef __attribute__((ext_vector_type(8)))  float  v8f;

#define B_       8
#define C_       64
#define H_       128
#define W_       128
#define K2_      9
#define KC       576            // C_ * K2_
#define KC_PAD   584            // padded LDS row stride (halves), spreads banks
#define NPIX     64             // pixels per block
#define NTHREADS 256            // 8 wave32
#define KSTEPS   (KC / 32)      // 18 WMMA K-steps

// LDS layout (bytes):
//   [0)              vals : 64 rows x KC_PAD bf16 (im2col, then deform-sampled vals)
//   [VALS_BYTES)     offb : 27 x 64 f32           (dy / dx / sigmoid(mask))
//   [WREG_OFF)       wreg : 64 rows x KC_PAD bf16 (phase1: 32-row w_off; phase3: w)
#define VALS_BYTES   (NPIX * KC_PAD * 2)          // 74752
#define OFFB_BYTES   (27 * NPIX * 4)              // 6912
#define WREG_OFF     (VALS_BYTES + OFFB_BYTES)    // 81664 (16B aligned)
#define WREG_BYTES   (64 * KC_PAD * 2)            // 74752
#define SMEM_BYTES   (WREG_OFF + WREG_BYTES)      // 156416  (<= 160KB -> 2 WG / WGP)

// bf16 weight pack (in workspace, after y1):
#define Y1_BYTES     ((size_t)B_ * C_ * H_ * W_ * 4)   // 33,554,432
#define OFF_WOFF1    0
#define OFF_W1       (32 * KC)                          // 18432
#define OFF_WOFF2    (OFF_W1 + 64 * KC)                 // 55296
#define OFF_W2       (OFF_WOFF2 + 32 * KC)              // 73728

__device__ __forceinline__ int iclamp(int v, int lo, int hi) {
    return v < lo ? lo : (v > hi ? hi : v);
}

// Low 32 bits of a generic pointer to __shared__ data == LDS byte address
// (the ISA maps an LDS-aperture address to its low 32 bits).
__device__ __forceinline__ unsigned lds_addr_of(const void* p) {
    return (unsigned)(unsigned long long)p;
}

// Async DMA: 16 bytes global -> LDS, tracked by ASYNCcnt (CDNA5 path).
__device__ __forceinline__ void async_copy_b128(unsigned lds_byte_addr,
                                                const void* gptr) {
    asm volatile("global_load_async_to_lds_b128 %0, %1, off"
                 :: "v"(lds_byte_addr), "v"((unsigned long long)gptr)
                 : "memory");
}

__device__ __forceinline__ void wait_asynccnt0() {
    asm volatile("s_wait_asynccnt 0x0" ::: "memory");
}

__device__ __forceinline__ v16bf make_a_frag(const __bf16* row, int kb, int h) {
    // A-matrix 16x32 bf16 (contiguous K, matching B): lanes 0-15 hold K {kb..kb+15},
    // lanes 16-31 hold K {kb+16..kb+31}. h = lane>>4.
    v8bf a0 = *(const v8bf*)(row + kb + h * 16);
    v8bf a1 = *(const v8bf*)(row + kb + h * 16 + 8);
    return __builtin_shufflevector(a0, a1, 0, 1, 2, 3, 4, 5, 6, 7,
                                           8, 9, 10, 11, 12, 13, 14, 15);
}

// ---------------------------------------------------------------------------
// One-shot weight pack: fp32 -> bf16, offset weights padded 27->32 rows.
// ---------------------------------------------------------------------------
__global__ __launch_bounds__(NTHREADS)
void pack_weights_kernel(const float* __restrict__ w_off1,
                         const float* __restrict__ w1,
                         const float* __restrict__ w_off2,
                         const float* __restrict__ w2,
                         __bf16* __restrict__ pack)
{
    const int stride = gridDim.x * NTHREADS;
    const int t0 = blockIdx.x * NTHREADS + threadIdx.x;
    for (int i = t0; i < 32 * KC; i += stride) {
        const int r = i / KC;
        pack[OFF_WOFF1 + i] = (r < 27) ? (__bf16)w_off1[i] : (__bf16)0.0f;
        pack[OFF_WOFF2 + i] = (r < 27) ? (__bf16)w_off2[i] : (__bf16)0.0f;
    }
    for (int i = t0; i < 64 * KC; i += stride) {
        pack[OFF_W1 + i] = (__bf16)w1[i];
        pack[OFF_W2 + i] = (__bf16)w2[i];
    }
}

// ---------------------------------------------------------------------------
// Fused DCN layer: offset conv (WMMA) -> bilinear sample -> main conv (WMMA).
// ---------------------------------------------------------------------------
__global__ __launch_bounds__(NTHREADS)
void dcn_layer_kernel(const float*  __restrict__ src,     // (B,C,H,W)
                      const float*  __restrict__ resid,   // x or nullptr
                      float*        __restrict__ dst,     // (B,C,H,W)
                      const __bf16* __restrict__ woffP,   // packed 32 x 576 bf16
                      const float*  __restrict__ b_off,   // (27)
                      const __bf16* __restrict__ wmainP,  // packed 64 x 576 bf16
                      const float*  __restrict__ bias,    // (C)
                      int do_relu)
{
    __shared__ __align__(16) char smem[SMEM_BYTES];
    __bf16* vals = (__bf16*)smem;                       // [NPIX][KC_PAD]
    float*  offb = (float*)(smem + VALS_BYTES);         // [27][NPIX]
    __bf16* wreg = (__bf16*)(smem + WREG_OFF);          // [<=64][KC_PAD]
    const unsigned wreg_lds = lds_addr_of(wreg);

    const int tid  = threadIdx.x;
    const int lane = tid & 31;
    const int wv   = tid >> 5;

    // 256 blocks per image: 128 rows x 2 x-tiles of 64
    const int blk   = blockIdx.x;
    const int bIdx  = blk >> 8;
    const int rem   = blk & 255;
    const int yy    = rem >> 1;
    const int xbase = (rem & 1) * NPIX;

    const float* srcB = src + (size_t)bIdx * C_ * H_ * W_;

    // ------- Phase 1: async-stage w_off to LDS, overlap with im2col --------
    // 32 rows x 576 bf16: 72 x b128 chunks per row, 2304 total -> 9 / thread.
    for (int j = tid; j < 32 * (KC / 8); j += NTHREADS) {
        const int r  = j / (KC / 8);
        const int c8 = j - r * (KC / 8);
        async_copy_b128(wreg_lds + (unsigned)(r * KC_PAD + c8 * 8) * 2u,
                        woffP + r * KC + c8 * 8);
    }
    {
        const int p  = tid & (NPIX - 1);
        const int g  = tid >> 6;                 // 0..3 channel group
        const int xx = xbase + p;
        for (int c = g; c < C_; c += 4) {
            const float* pc = srcB + c * H_ * W_;
            #pragma unroll
            for (int k = 0; k < K2_; ++k) {
                const int ys = yy + (k / 3) - 1;
                const int xs = xx + (k % 3) - 1;
                float v = 0.0f;
                if (ys >= 0 && ys < H_ && xs >= 0 && xs < W_) v = pc[ys * W_ + xs];
                vals[p * KC_PAD + c * K2_ + k] = (__bf16)v;
            }
        }
    }
    wait_asynccnt0();
    __syncthreads();

    // ------------- Phase 2: offset GEMM 32x576 @ 576x64, all 8 waves -------
    {
        const int mrow = wv & 1;
        const int ncol = wv >> 1;                // 0..3
        const int m = mrow * 16 + (lane & 15);
        const int h = lane >> 4;
        const int n = ncol * 16 + (lane & 15);
        const __bf16* arow = wreg + m * KC_PAD;
        const __bf16* brow = vals + n * KC_PAD;
        v8f acc = {};
        #pragma unroll 2
        for (int kk = 0; kk < KSTEPS; ++kk) {
            const int kb = kk * 32;
            v16bf a  = make_a_frag(arow, kb, h);
            v16bf bm = *(const v16bf*)(brow + kb + h * 16);  // 16 consecutive K
            acc = __builtin_amdgcn_wmma_f32_16x16x32_bf16(false, a, false, bm,
                                                          (short)0, acc, false, false);
        }
        // D layout: VGPR r -> M = r (+8 for lanes 16-31), N = lane & 15
        const int pp = ncol * 16 + (lane & 15);
        const int mb = mrow * 16 + ((lane >> 4) ? 8 : 0);
        #pragma unroll
        for (int r = 0; r < 8; ++r) {
            const int o = mb + r;
            if (o < 27) {
                float v = acc[r] + b_off[o];
                if (o >= 18) {                     // sigmoid via v_exp + v_rcp
                    v = __builtin_amdgcn_rcpf(1.0f + __expf(-v));
                }
                offb[o * NPIX + pp] = v;
            }
        }
    }
    __syncthreads();

    // --- Phase 3: async-stage main weights, overlap with bilinear sampling --
    for (int j = tid; j < 64 * (KC / 8); j += NTHREADS) {   // 18 / thread
        const int r  = j / (KC / 8);
        const int c8 = j - r * (KC / 8);
        async_copy_b128(wreg_lds + (unsigned)(r * KC_PAD + c8 * 8) * 2u,
                        wmainP + r * KC + c8 * 8);
    }
    {
        const int p  = tid & (NPIX - 1);
        const int g  = tid >> 6;
        const int xx = xbase + p;
        #pragma unroll
        for (int k = 0; k < K2_; ++k) {
            const float dyv = offb[(0       + k) * NPIX + p];
            const float dxv = offb[(K2_     + k) * NPIX + p];
            const float mv  = offb[(2 * K2_ + k) * NPIX + p];
            const float py = (float)yy + (float)(k / 3 - 1) + dyv;
            const float px = (float)xx + (float)(k % 3 - 1) + dxv;
            const float fy0 = floorf(py), fx0 = floorf(px);
            const int   y0 = (int)fy0,   x0 = (int)fx0;
            const float wy = py - fy0,   wx = px - fx0;
            const int   y1 = y0 + 1,     x1 = x0 + 1;
            const float vy0 = (y0 >= 0 && y0 < H_) ? 1.0f : 0.0f;
            const float vy1 = (y1 >= 0 && y1 < H_) ? 1.0f : 0.0f;
            const float vx0 = (x0 >= 0 && x0 < W_) ? 1.0f : 0.0f;
            const float vx1 = (x1 >= 0 && x1 < W_) ? 1.0f : 0.0f;
            const int y0c = iclamp(y0, 0, H_ - 1), y1c = iclamp(y1, 0, H_ - 1);
            const int x0c = iclamp(x0, 0, W_ - 1), x1c = iclamp(x1, 0, W_ - 1);
            const float w00 = (1.0f - wy) * (1.0f - wx) * vy0 * vx0 * mv;
            const float w01 = (1.0f - wy) * wx          * vy0 * vx1 * mv;
            const float w10 = wy * (1.0f - wx)          * vy1 * vx0 * mv;
            const float w11 = wy * wx                   * vy1 * vx1 * mv;
            const int i00 = y0c * W_ + x0c, i01 = y0c * W_ + x1c;
            const int i10 = y1c * W_ + x0c, i11 = y1c * W_ + x1c;
            for (int c = g; c < C_; c += 4) {
                const float* pc = srcB + c * H_ * W_;
                const float v = w00 * pc[i00] + w01 * pc[i01]
                              + w10 * pc[i10] + w11 * pc[i11];
                vals[p * KC_PAD + c * K2_ + k] = (__bf16)v;
            }
        }
    }
    wait_asynccnt0();
    __syncthreads();

    // ---------------- Phase 4: main GEMM 64x576 @ 576x64 -------------------
    // 16 output tiles; each wave owns (mrow, nc0) and (mrow, nc0+1),
    // A-fragment reused across the two N-tiles.
    {
        const int mrow = wv & 3;
        const int nc0  = (wv >> 2) * 2;          // 0 or 2
        const int m  = mrow * 16 + (lane & 15);
        const int h  = lane >> 4;
        const int n0 = nc0 * 16 + (lane & 15);
        const __bf16* arow  = wreg + m * KC_PAD;
        const __bf16* brow0 = vals + n0 * KC_PAD;
        const __bf16* brow1 = brow0 + 16 * KC_PAD;
        v8f acc0 = {}, acc1 = {};
        #pragma unroll 2
        for (int kk = 0; kk < KSTEPS; ++kk) {
            const int kb = kk * 32;
            v16bf a  = make_a_frag(arow, kb, h);
            v16bf b0 = *(const v16bf*)(brow0 + kb + h * 16);
            v16bf b1 = *(const v16bf*)(brow1 + kb + h * 16);
            acc0 = __builtin_amdgcn_wmma_f32_16x16x32_bf16(false, a, false, b0,
                                                           (short)0, acc0, false, false);
            acc1 = __builtin_amdgcn_wmma_f32_16x16x32_bf16(false, a, false, b1,
                                                           (short)0, acc1, false, false);
        }
        const int mb  = mrow * 16 + ((lane >> 4) ? 8 : 0);
        #pragma unroll
        for (int t = 0; t < 2; ++t) {
            const int pp = (nc0 + t) * 16 + (lane & 15);
            const int xx = xbase + pp;
            const v8f& acc = t ? acc1 : acc0;
            #pragma unroll
            for (int r = 0; r < 8; ++r) {
                const int o = mb + r;
                const size_t idx = (((size_t)bIdx * C_ + o) * H_ + yy) * W_ + xx;
                float v = acc[r] + bias[o];
                if (do_relu) v = fmaxf(v, 0.0f);
                if (resid)   v += resid[idx];
                dst[idx] = v;
            }
        }
    }
}

extern "C" void kernel_launch(void* const* d_in, const int* in_sizes, int n_in,
                              void* d_out, int out_size, void* d_ws, size_t ws_size,
                              hipStream_t stream) {
    const float* x      = (const float*)d_in[0];
    const float* w_off1 = (const float*)d_in[1];
    const float* b_off1 = (const float*)d_in[2];
    const float* w1     = (const float*)d_in[3];
    const float* b1     = (const float*)d_in[4];
    const float* w_off2 = (const float*)d_in[5];
    const float* b_off2 = (const float*)d_in[6];
    const float* w2     = (const float*)d_in[7];
    const float* b2     = (const float*)d_in[8];
    float* out = (float*)d_out;

    char*   ws   = (char*)d_ws;
    float*  y1   = (float*)ws;                    // 33.5 MB intermediate
    __bf16* pack = (__bf16*)(ws + Y1_BYTES);      // 221,184 B packed weights

    // Pack all weights to bf16 once (deterministic, cheap).
    pack_weights_kernel<<<64, NTHREADS, 0, stream>>>(w_off1, w1, w_off2, w2, pack);

    const int nblocks = B_ * H_ * (W_ / NPIX);    // 2048

    // Layer 1: y1 = relu(dcn(x, w_off1, b_off1, w1, b1))
    dcn_layer_kernel<<<nblocks, NTHREADS, 0, stream>>>(
        x, nullptr, y1, pack + OFF_WOFF1, b_off1, pack + OFF_W1, b1, /*do_relu=*/1);

    // Layer 2: out = dcn(y1, w_off2, b_off2, w2, b2) + x
    dcn_layer_kernel<<<nblocks, NTHREADS, 0, stream>>>(
        y1, x, out, pack + OFF_WOFF2, b_off2, pack + OFF_W2, b2, /*do_relu=*/0);
}
